// RC_Explainer_Batch_30339648979128
// MI455X (gfx1250) — compile-verified, matrix operand order checked by reading, not run on hardware
//
#include <hip/hip_runtime.h>
#include <hip/hip_bf16.h>

#define N_NODES 50000
#define N_EDGES 262144
#define N_OCP   32768
#define E_AVA   (N_EDGES - N_OCP)   /* 229376 */
#define N_GRAPHS 64
#define HD 128
#define CHUNK 32768
#define NCHUNK (E_AVA / CHUNK)      /* 7 */

typedef __attribute__((ext_vector_type(16))) __bf16        bf16x16;
typedef __attribute__((ext_vector_type(8)))  float         floatx8;
typedef __attribute__((ext_vector_type(4)))  unsigned int  uintx4;
typedef __attribute__((ext_vector_type(4)))  float         floatx4;

union FragU { uintx4 q[2]; bf16x16 v; };

#if defined(__gfx1250__) && __has_builtin(__builtin_amdgcn_global_load_async_to_lds_b128) && __has_builtin(__builtin_amdgcn_s_wait_asynccnt)
#define USE_ASYNC_COPY 1
typedef int gv4i __attribute__((vector_size(16)));
typedef gv4i __attribute__((address_space(1)))* as1_v4i_ptr;
typedef gv4i __attribute__((address_space(3)))* as3_v4i_ptr;
__device__ __forceinline__ void async_cp16(const void* g, void* l) {
  __builtin_amdgcn_global_load_async_to_lds_b128(
      (as1_v4i_ptr)g, (as3_v4i_ptr)l, 0, 0);
}
#else
#define USE_ASYNC_COPY 0
#endif

__device__ __forceinline__ unsigned short f2bf(float f) {
  unsigned u = __float_as_uint(f);
  u += 0x7FFFu + ((u >> 16) & 1u);
  return (unsigned short)(u >> 16);
}
__device__ __forceinline__ float bf2f(unsigned short h) {
  return __uint_as_float(((unsigned)h) << 16);
}
__device__ __forceinline__ float eluf(float x) { return x > 0.f ? x : (expf(x) - 1.f); }
__device__ __forceinline__ unsigned fkey(float f) {           // monotone float->uint
  unsigned u = __float_as_uint(f);
  return (u & 0x80000000u) ? ~u : (u | 0x80000000u);
}
__device__ __forceinline__ float funkey(unsigned k) {
  return __uint_as_float((k & 0x80000000u) ? (k ^ 0x80000000u) : ~k);
}

// ---------------------------------------------------------------------------
// Generic bf16 WMMA GEMM:  C[M,N] = op(A[M,K] @ Bt[N,K]^T + bias)
// MODE 0: A is a dense bf16 matrix.
// MODE 1: A row r = concat(ava[sidx[r]], ava[didx[r]])  (K must be 256).
// Block tile 128x128, 8 waves (4x2), wave tile 32x64, K-step 32.
// N must be a multiple of 128.
// ---------------------------------------------------------------------------
template<int MODE, bool ELU, bool BIAS, bool OF32>
__global__ __launch_bounds__(256)
void gemm_wmma_k(const unsigned short* __restrict__ A,
                 const unsigned short* __restrict__ ava,
                 const int* __restrict__ sidx,
                 const int* __restrict__ didx,
                 const unsigned short* __restrict__ Bt,
                 const float* __restrict__ bias,
                 void* __restrict__ Cout,
                 int M, int N, int K)
{
  __shared__ unsigned short As[128 * 40];   // 80B pitch: conflict-free b128 reads
  __shared__ unsigned short Bs[128 * 40];

  const int tid  = threadIdx.x;
  const int lane = tid & 31;
  const int wave = tid >> 5;
  const int wm   = wave >> 1;               // 0..3 : 32-row strip
  const int wn   = wave & 1;                // 0..1 : 64-col strip
  const int mBase = blockIdx.x * 128;
  const int nBase = blockIdx.y * 128;

  floatx8 acc[2][4] = {};                   // 32x64 per wave

  const int lrow = tid >> 1, lseg = tid & 1;  // loader: 128 rows x 2 x 16 halves

  for (int k0 = 0; k0 < K; k0 += 32) {
    // ---- stage A tile (optional gather/concat; zero-fill ragged tail) ----
    {
      const int gRow = mBase + lrow;
      unsigned short* ldsA = &As[lrow * 40 + lseg * 16];
#if USE_ASYNC_COPY
      if (MODE == 1) {                       // gather rows, always in-range
        const int col  = k0 + lseg * 16;
        const int node = (col < HD) ? sidx[gRow] : didx[gRow];
        const unsigned short* srcp = ava + (size_t)node * HD + (col & (HD - 1));
        async_cp16(srcp, ldsA);
        async_cp16(srcp + 8, ldsA + 8);
      } else if (mBase + 128 <= M) {         // full block, direct rows
        const unsigned short* srcp = A + (size_t)gRow * K + k0 + lseg * 16;
        async_cp16(srcp, ldsA);
        async_cp16(srcp + 8, ldsA + 8);
      } else
#endif
      {
        uintx4 av0 = {0,0,0,0}, av1 = {0,0,0,0};
        if (gRow < M) {
          const unsigned short* srcp;
          if (MODE == 0) {
            srcp = A + (size_t)gRow * K + k0 + lseg * 16;
          } else {
            const int col  = k0 + lseg * 16;
            const int node = (col < HD) ? sidx[gRow] : didx[gRow];
            srcp = ava + (size_t)node * HD + (col & (HD - 1));
          }
          const uintx4* s4 = (const uintx4*)srcp;
          av0 = s4[0]; av1 = s4[1];
        }
        uintx4* d4 = (uintx4*)ldsA;
        d4[0] = av0; d4[1] = av1;
      }
    }
    // ---- stage B tile (weights, transposed [N][K]; always in-range) ----
    {
      const unsigned short* gB = Bt + (size_t)(nBase + lrow) * K + k0 + lseg * 16;
      unsigned short* ldsB = &Bs[lrow * 40 + lseg * 16];
#if USE_ASYNC_COPY
      async_cp16(gB, ldsB);
      async_cp16(gB + 8, ldsB + 8);
#else
      const uintx4* s4 = (const uintx4*)gB;
      uintx4* b4 = (uintx4*)ldsB;
      b4[0] = s4[0]; b4[1] = s4[1];
#endif
    }
#if USE_ASYNC_COPY
    __builtin_amdgcn_s_wait_asynccnt(0);
#endif
    __syncthreads();

    // ---- fragments per ISA 16-bit layout:
    // lane<16: K 0-7 & 16-23 ; lane>=16: K 8-15 & 24-31
    const int r  = lane & 15;
    const int qi = lane >> 4;                 // 0/1 -> uintx4 index 0/1 (+2)
    FragU a[2], b[4];
#pragma unroll
    for (int mi = 0; mi < 2; ++mi) {
      const uintx4* p = (const uintx4*)&As[(wm * 32 + mi * 16 + r) * 40];
      a[mi].q[0] = p[qi]; a[mi].q[1] = p[qi + 2];
    }
#pragma unroll
    for (int ni = 0; ni < 4; ++ni) {
      const uintx4* p = (const uintx4*)&Bs[(wn * 64 + ni * 16 + r) * 40];
      b[ni].q[0] = p[qi]; b[ni].q[1] = p[qi + 2];
    }
#pragma unroll
    for (int mi = 0; mi < 2; ++mi)
#pragma unroll
      for (int ni = 0; ni < 4; ++ni)
        acc[mi][ni] = __builtin_amdgcn_wmma_f32_16x16x32_bf16(
            false, a[mi].v, false, b[ni].v, (short)0, acc[mi][ni], false, false);
    __syncthreads();
  }

  // ---- epilogue: C/D layout = lanes 0-15 N=lane M=g ; lanes 16-31 N=lane-16 M=g+8
  const int r  = lane & 15;
  const int mo = (lane >> 4) * 8;
#pragma unroll
  for (int mi = 0; mi < 2; ++mi)
#pragma unroll
    for (int ni = 0; ni < 4; ++ni) {
      const int n  = nBase + wn * 64 + ni * 16 + r;
      const int mb = mBase + wm * 32 + mi * 16 + mo;
      const float bv = BIAS ? bias[n] : 0.f;
      floatx8 c = acc[mi][ni];
#pragma unroll
      for (int g = 0; g < 8; ++g) {
        const int m = mb + g;
        if (m >= M) continue;
        float v = c[g] + bv;
        if (ELU) v = eluf(v);
        if (OF32) ((float*)Cout)[(size_t)m * N + n] = v;
        else      ((unsigned short*)Cout)[(size_t)m * N + n] = f2bf(v);
      }
    }
}

// ---------------------------------------------------------------------------
// Small helper kernels
// ---------------------------------------------------------------------------
__global__ void cvt_bf16_k(const float* __restrict__ in, unsigned short* __restrict__ out, int n) {
  int i = blockIdx.x * 256 + threadIdx.x;
  if (i < n) out[i] = f2bf(in[i]);
}

__global__ void transpose_w_k(const float* __restrict__ W, unsigned short* __restrict__ Wt, int K, int N) {
  int i = blockIdx.x * 256 + threadIdx.x;
  if (i < K * N) { int k = i / N, n = i - k * N; Wt[(size_t)n * K + k] = f2bf(W[i]); }
}

// edge scatter: agg_full[dst] += M[src]; agg_ocp too for first N_OCP edges
__global__ void scatter_k(const float* __restrict__ Mn, const int* __restrict__ src,
                          const int* __restrict__ dst,
                          float* __restrict__ aggF, float* __restrict__ aggO) {
  int gid = blockIdx.x * 256 + threadIdx.x;
  int e = gid >> 5;
  if (e >= N_EDGES) return;
  int c = (gid & 31) * 4;
  floatx4 v = *(const floatx4*)(Mn + (size_t)src[e] * HD + c);
  float* f = aggF + (size_t)dst[e] * HD + c;
  atomicAdd(f + 0, v.x); atomicAdd(f + 1, v.y); atomicAdd(f + 2, v.z); atomicAdd(f + 3, v.w);
  if (e < N_OCP) {
    float* o = aggO + (size_t)dst[e] * HD + c;
    atomicAdd(o + 0, v.x); atomicAdd(o + 1, v.y); atomicAdd(o + 2, v.z); atomicAdd(o + 3, v.w);
  }
}

__global__ void ava_k(const float* __restrict__ base, const float* __restrict__ aggF,
                      const float* __restrict__ aggO, unsigned short* __restrict__ ava, int n) {
  int i = blockIdx.x * 256 + threadIdx.x;
  if (i < n) ava[i] = f2bf(eluf(base[i] + aggF[i]) - eluf(base[i] + aggO[i]));
}

// sel[gi] = h[i,:] . Wp2[:, y[batch[src]]] + bp2[...] ; also stores seg
__global__ void sel_k(const unsigned short* __restrict__ h, const int* __restrict__ ei0,
                      const int* __restrict__ batch, const int* __restrict__ y,
                      const float* __restrict__ Wp2, const float* __restrict__ bp2,
                      float* __restrict__ sel, int* __restrict__ seg, int rowBase) {
  int i = blockIdx.x * 256 + threadIdx.x;
  if (i >= CHUNK) return;
  int gi = rowBase + i;
  int s  = ei0[N_OCP + gi];
  int g  = batch[s];
  int yy = y[g];
  const unsigned short* hr = h + (size_t)i * HD;
  float acc = bp2[yy];
#pragma unroll 4
  for (int c = 0; c < HD; ++c) acc += bf2f(hr[c]) * Wp2[c * 10 + yy];
  sel[gi] = acc;
  seg[gi] = g;
}

__global__ void init_pg_k(unsigned* mxkey, float* sum, unsigned* pmk, int* amin) {
  int g = threadIdx.x;
  if (g < N_GRAPHS) { mxkey[g] = 0u; sum[g] = 0.f; pmk[g] = 0u; amin[g] = 0x7fffffff; }
}
__global__ void segmax_k(const float* __restrict__ sel, const int* __restrict__ seg, unsigned* mxkey) {
  int i = blockIdx.x * 256 + threadIdx.x;
  if (i < E_AVA) atomicMax(&mxkey[seg[i]], fkey(sel[i]));
}
__global__ void exsum_k(const float* __restrict__ sel, const int* __restrict__ seg,
                        const unsigned* __restrict__ mxkey, float* __restrict__ ex, float* sum) {
  int i = blockIdx.x * 256 + threadIdx.x;
  if (i >= E_AVA) return;
  float e = expf(sel[i] - funkey(mxkey[seg[i]]));
  ex[i] = e;
  atomicAdd(&sum[seg[i]], e);
}
__global__ void probs_k(const float* __restrict__ ex, const int* __restrict__ seg,
                        const float* __restrict__ sum, float* __restrict__ out, unsigned* pmk) {
  int i = blockIdx.x * 256 + threadIdx.x;
  if (i >= E_AVA) return;
  float p = ex[i] / sum[seg[i]];
  out[i] = p;
  atomicMax(&pmk[seg[i]], fkey(p));
}
__global__ void amin_k(const float* __restrict__ out, const int* __restrict__ seg,
                       const unsigned* __restrict__ pmk, int* amin) {
  int i = blockIdx.x * 256 + threadIdx.x;
  if (i >= E_AVA) return;
  int idx = (out[i] == funkey(pmk[seg[i]])) ? i : E_AVA;
  atomicMin(&amin[seg[i]], idx);
}
__global__ void fin_k(const unsigned* __restrict__ pmk, const int* __restrict__ amin,
                      float* __restrict__ out) {
  int g = threadIdx.x;
  if (g < N_GRAPHS) {
    out[E_AVA + g]            = funkey(pmk[g]);
    out[E_AVA + N_GRAPHS + g] = (float)amin[g];
  }
}

// ---------------------------------------------------------------------------
extern "C" void kernel_launch(void* const* d_in, const int* in_sizes, int n_in,
                              void* d_out, int out_size, void* d_ws, size_t ws_size,
                              hipStream_t stream) {
  const float* x      = (const float*)d_in[0];
  const int*   ei     = (const int*)d_in[1];
  const int*   batch  = (const int*)d_in[2];
  const int*   y      = (const int*)d_in[3];
  // d_in[4] = n_ocp scalar (device); compile-time N_OCP matches reference
  const float* W_self = (const float*)d_in[5];
  const float* W_nbr  = (const float*)d_in[6];
  const float* b_gnn  = (const float*)d_in[7];
  const float* Wr1 = (const float*)d_in[8];   const float* br1 = (const float*)d_in[9];
  const float* Wr2 = (const float*)d_in[10];  const float* br2 = (const float*)d_in[11];
  const float* Wr3 = (const float*)d_in[12];  const float* br3 = (const float*)d_in[13];
  const float* Wp1 = (const float*)d_in[14];  const float* bp1 = (const float*)d_in[15];
  const float* Wp2 = (const float*)d_in[16];  const float* bp2 = (const float*)d_in[17];
  const int* ei0 = ei;
  const int* ei1 = ei + N_EDGES;
  float* out = (float*)d_out;

  // ---- workspace carve-out (~200 MB) ----
  unsigned char* w = (unsigned char*)d_ws;
  size_t off = 0;
  auto take = [&](size_t bytes) -> void* {
    void* p = w + off; off += (bytes + 255) & ~(size_t)255; return p;
  };
  unsigned short* xb   = (unsigned short*)take((size_t)N_NODES * HD * 2);
  unsigned short* Wst  = (unsigned short*)take(128 * 128 * 2);
  unsigned short* Wnt  = (unsigned short*)take(128 * 128 * 2);
  unsigned short* Wr1t = (unsigned short*)take(512 * 256 * 2);
  unsigned short* Wr2t = (unsigned short*)take(256 * 512 * 2);
  unsigned short* Wr3t = (unsigned short*)take(128 * 256 * 2);
  unsigned short* Wp1t = (unsigned short*)take(128 * 128 * 2);
  float* base = (float*)take((size_t)N_NODES * HD * 4);
  float* Mn   = (float*)take((size_t)N_NODES * HD * 4);
  float* aggF = (float*)take((size_t)N_NODES * HD * 4);
  float* aggO = (float*)take((size_t)N_NODES * HD * 4);
  unsigned short* avab = (unsigned short*)take((size_t)N_NODES * HD * 2);
  unsigned short* actA = (unsigned short*)take((size_t)CHUNK * 512 * 2);
  unsigned short* actB = (unsigned short*)take((size_t)CHUNK * 512 * 2);
  float* sel = (float*)take((size_t)E_AVA * 4);
  float* ex  = (float*)take((size_t)E_AVA * 4);
  int*   seg = (int*)take((size_t)E_AVA * 4);
  unsigned* mxkey = (unsigned*)take(256);
  float*    sum   = (float*)take(256);
  unsigned* pmk   = (unsigned*)take(256);
  int*      amin  = (int*)take(256);

  // ---- precision conversion / weight transposes ----
  cvt_bf16_k<<<(N_NODES * HD + 255) / 256, 256, 0, stream>>>(x, xb, N_NODES * HD);
  transpose_w_k<<<(128 * 128 + 255) / 256, 256, 0, stream>>>(W_self, Wst, 128, 128);
  transpose_w_k<<<(128 * 128 + 255) / 256, 256, 0, stream>>>(W_nbr,  Wnt, 128, 128);
  transpose_w_k<<<(256 * 512 + 255) / 256, 256, 0, stream>>>(Wr1, Wr1t, 256, 512);
  transpose_w_k<<<(512 * 256 + 255) / 256, 256, 0, stream>>>(Wr2, Wr2t, 512, 256);
  transpose_w_k<<<(256 * 128 + 255) / 256, 256, 0, stream>>>(Wr3, Wr3t, 256, 128);
  transpose_w_k<<<(128 * 128 + 255) / 256, 256, 0, stream>>>(Wp1, Wp1t, 128, 128);

  // ---- node phase: base = x@W_self + b ; Mn = x@W_nbr ----
  dim3 ngrid((N_NODES + 127) / 128, 1);
  gemm_wmma_k<0, false, true,  true><<<ngrid, 256, 0, stream>>>(
      xb, nullptr, nullptr, nullptr, Wst, b_gnn, base, N_NODES, 128, 128);
  gemm_wmma_k<0, false, false, true><<<ngrid, 256, 0, stream>>>(
      xb, nullptr, nullptr, nullptr, Wnt, nullptr, Mn, N_NODES, 128, 128);

  (void)hipMemsetAsync(aggF, 0, (size_t)N_NODES * HD * 4, stream);
  (void)hipMemsetAsync(aggO, 0, (size_t)N_NODES * HD * 4, stream);
  scatter_k<<<(N_EDGES * 32) / 256, 256, 0, stream>>>(Mn, ei0, ei1, aggF, aggO);
  ava_k<<<(N_NODES * HD + 255) / 256, 256, 0, stream>>>(base, aggF, aggO, avab, N_NODES * HD);

  init_pg_k<<<1, 64, 0, stream>>>(mxkey, sum, pmk, amin);

  // ---- edge MLP, chunked over 7 x 32768 rows ----
  for (int c = 0; c < NCHUNK; ++c) {
    const int rowBase = c * CHUNK;
    // L1: concat(ava[src],ava[dst]) [*,256] @ [256,512] + br1, elu
    gemm_wmma_k<1, true,  true, false><<<dim3(CHUNK / 128, 4), 256, 0, stream>>>(
        nullptr, avab, ei0 + N_OCP + rowBase, ei1 + N_OCP + rowBase,
        Wr1t, br1, actA, CHUNK, 512, 256);
    // L2: [*,512] @ [512,256] + br2, elu
    gemm_wmma_k<0, true,  true, false><<<dim3(CHUNK / 128, 2), 256, 0, stream>>>(
        actA, nullptr, nullptr, nullptr, Wr2t, br2, actB, CHUNK, 256, 512);
    // L3: [*,256] @ [256,128] + br3 (no act)
    gemm_wmma_k<0, false, true, false><<<dim3(CHUNK / 128, 1), 256, 0, stream>>>(
        actB, nullptr, nullptr, nullptr, Wr3t, br3, actA, CHUNK, 128, 256);
    // L4: h = elu([*,128] @ [128,128] + bp1)
    gemm_wmma_k<0, true,  true, false><<<dim3(CHUNK / 128, 1), 256, 0, stream>>>(
        actA, nullptr, nullptr, nullptr, Wp1t, bp1, actB, CHUNK, 128, 128);
    // sel = h . Wp2[:, y_edge] + bp2[y_edge]
    sel_k<<<CHUNK / 256, 256, 0, stream>>>(actB, ei0, batch, y, Wp2, bp2, sel, seg, rowBase);
  }

  // ---- segmented softmax-max / argmax ----
  const int EB = (E_AVA + 255) / 256;
  segmax_k<<<EB, 256, 0, stream>>>(sel, seg, mxkey);
  exsum_k <<<EB, 256, 0, stream>>>(sel, seg, mxkey, ex, sum);
  probs_k <<<EB, 256, 0, stream>>>(ex, seg, sum, out, pmk);
  amin_k  <<<EB, 256, 0, stream>>>(out, seg, pmk, amin);
  fin_k   <<<1, 64, 0, stream>>>(pmk, amin, out);
}
